// selfAttention_prompt_69595650064831
// MI455X (gfx1250) — matrix operand order repaired and checked by reading, not compiled, not run on hardware
//
#include <hip/hip_runtime.h>

typedef _Float16 h8  __attribute__((ext_vector_type(8)));
typedef _Float16 h16 __attribute__((ext_vector_type(16)));
typedef float    f8  __attribute__((ext_vector_type(8)));
typedef int      v4i __attribute__((ext_vector_type(4)));

#define WMMA_F32_F16(a, b, c) \
  __builtin_amdgcn_wmma_f32_16x16x32_f16(false, (a), false, (b), (short)0, (c), false, false)

#define BB 4
#define SS 2048
#define DD 256
#define HH 8
#define HD 2048           // h * d
#define MM (BB * SS)      // 8192 rows

// padded LDS strides (halves) -> gap-4 bank stride for 16B fragment reads
#define KSTR 264          // 32-key K tile rows
#define VSTR 40           // 256 d-rows of V tile
#define PSTR 40           // P tile rows

#if __has_builtin(__builtin_amdgcn_global_load_async_to_lds_b128)
#define HAVE_ASYNC_LDS 1
#else
#define HAVE_ASYNC_LDS 0
#endif

typedef __attribute__((address_space(1))) v4i* g_v4i_ptr;  // global b128 element
typedef __attribute__((address_space(3))) v4i* l_v4i_ptr;  // LDS b128 element

__device__ __forceinline__ h16 ld_h16(const _Float16* p) {
  return *(const h16*)p;
}
// A-fragment style load: two contiguous 8-half (16B) pieces
__device__ __forceinline__ h16 ld_h16_split(const _Float16* p0, const _Float16* p1) {
  union { h16 v; h8 h[2]; } u;
  u.h[0] = *(const h8*)p0;
  u.h[1] = *(const h8*)p1;
  return u.v;
}

// 16B global -> LDS staging element (async path when toolchain exposes it)
__device__ __forceinline__ void cp16B(const _Float16* g, _Float16* l) {
#if HAVE_ASYNC_LDS
  __builtin_amdgcn_global_load_async_to_lds_b128(
      (g_v4i_ptr)g, (l_v4i_ptr)l, 0, 0);
#else
  *(h8*)l = *(const h8*)g;
#endif
}
__device__ __forceinline__ void wait_async_stage() {
#if HAVE_ASYNC_LDS
#if __has_builtin(__builtin_amdgcn_s_wait_asynccnt)
  __builtin_amdgcn_s_wait_asynccnt(0);
#else
  asm volatile("s_wait_asynccnt 0x0" ::: "memory");
#endif
#endif
}

// ---------------------------------------------------------------- f32 -> f16
__global__ void cvt_f16_kernel(const float* __restrict__ s, _Float16* __restrict__ d, int n) {
  int i = blockIdx.x * blockDim.x + threadIdx.x;
  if (i < n) d[i] = (_Float16)s[i];
}

// ------------------------------------------------- QKV projection: C = X W^T + b
template <int TRANSPOSE_V>
__global__ __launch_bounds__(256) void proj_gemm_kernel(
    const _Float16* __restrict__ X, const _Float16* __restrict__ W,
    const float* __restrict__ bias, _Float16* __restrict__ out) {
  const int wave = threadIdx.x >> 5;
  const int lane = threadIdx.x & 31;
  const int r    = lane & 15;
  const int hi   = lane >> 4;
  const int row0  = blockIdx.x * 16;
  const int ncol0 = blockIdx.y * 128 + wave * 16;

  f8 acc = {};
  const _Float16* arow = X + (size_t)(row0 + r) * DD;
  const _Float16* brow = W + (size_t)(ncol0 + r) * DD;
#pragma unroll
  for (int c = 0; c < DD / 32; ++c) {
    h16 a = ld_h16_split(arow + c * 32 + hi * 8, arow + c * 32 + 16 + hi * 8);
    h16 b = ld_h16_split(brow + c * 32 + hi * 8, brow + c * 32 + 16 + hi * 8);
    acc = WMMA_F32_F16(a, b, acc);
  }
  const int   ncol = ncol0 + r;
  const float bv   = bias[ncol];
  if (TRANSPOSE_V) {
    const int bidx = row0 >> 11;
    const int s0   = row0 & (SS - 1);
    const int hcol = ncol >> 8;
    const int dcol = ncol & (DD - 1);
    h8 v;
#pragma unroll
    for (int j = 0; j < 8; ++j) v[j] = (_Float16)(acc[j] + bv);
    _Float16* dst = out + ((size_t)((bidx * HH + hcol) * DD + dcol)) * SS + s0 + hi * 8;
    *(h8*)dst = v;
  } else {
#pragma unroll
    for (int j = 0; j < 8; ++j)
      out[(size_t)(row0 + hi * 8 + j) * HD + ncol] = (_Float16)(acc[j] + bv);
  }
}

// --------------------------------------------------------- flash attention
// 4 waves / block, 64 query rows per (b,h) block; K/V tiles (32 keys) staged
// in double-buffered LDS (async global->LDS) and shared by all 4 waves.
__global__ __launch_bounds__(128) void attn_kernel(
    const _Float16* __restrict__ qb,   // [MM, HD]
    const _Float16* __restrict__ kb,   // [MM, HD]
    const _Float16* __restrict__ vt,   // [B, H, DD, SS]
    const int*      __restrict__ mask, // [B, SS]
    _Float16*       __restrict__ attn) // [MM, HD]
{
  const int h = blockIdx.y, b = blockIdx.z;
  const int t    = threadIdx.x;
  const int wave = t >> 5;
  const int lane = t & 31;
  const int r = lane & 15, hi = lane >> 4;
  const int row0 = blockIdx.x * 64 + wave * 16;

  __shared__ _Float16 kT[2][32][KSTR];   // keys x d (32x256 used)
  __shared__ _Float16 vT[2][DD][VSTR];   // d x keys (256x32 used)
  __shared__ _Float16 pT[4][16][PSTR];   // per-wave P bounce (16x32 used)

  const _Float16* kbase = kb + ((size_t)(b * SS)) * HD + h * DD;
  const _Float16* vbase = vt + ((size_t)(b * HH + h)) * DD * SS;
  const int*      mptr  = mask + b * SS;

  // resident Q A-fragments
  const _Float16* qbase = qb + ((size_t)(b * SS + row0 + r)) * HD + h * DD;
  h16 aq[8];
#pragma unroll
  for (int c = 0; c < 8; ++c)
    aq[c] = ld_h16_split(qbase + c * 32 + hi * 8, qbase + c * 32 + 16 + hi * 8);

  f8 o[16];
#pragma unroll
  for (int nc = 0; nc < 16; ++nc) o[nc] = (f8){};
  float mrow[8], lrow[8];
#pragma unroll
  for (int j = 0; j < 8; ++j) { mrow[j] = -1e9f; lrow[j] = 0.f; }

  // cooperative stage of one 32-key K tile + V tile into LDS buffer `buf`
  const int krow = t & 31;       // K: thread copies 64 halves of one key row
  const int kseg = t >> 5;
  const int d0   = t * 2;        // V: thread copies 2 d-rows x 32 halves
  auto stage = [&](int buf, int k0) {
    const _Float16* gk = kbase + (size_t)(k0 + krow) * HD + kseg * 64;
    _Float16*       lk = &kT[buf][krow][kseg * 64];
#pragma unroll
    for (int i = 0; i < 8; ++i) cp16B(gk + i * 8, lk + i * 8);
#pragma unroll
    for (int rr = 0; rr < 2; ++rr) {
      const _Float16* gv = vbase + (size_t)(d0 + rr) * SS + k0;
      _Float16*       lv = &vT[buf][d0 + rr][0];
#pragma unroll
      for (int i = 0; i < 4; ++i) cp16B(gv + i * 8, lv + i * 8);
    }
  };

  stage(0, 0);
  wait_async_stage();
  __syncthreads();

  for (int it = 0; it < SS / 32; ++it) {
    const int cur = it & 1;
    const int k0  = it * 32;
    if (it + 1 < SS / 32) stage(cur ^ 1, k0 + 32);   // overlap with compute

    // scores: two 16x16 tiles from LDS-resident K
    f8 s0 = {}, s1 = {};
#pragma unroll
    for (int c = 0; c < 8; ++c) {
      h16 bk0 = ld_h16_split(&kT[cur][r][c * 32 + hi * 8], &kT[cur][r][c * 32 + 16 + hi * 8]);
      h16 bk1 = ld_h16_split(&kT[cur][16 + r][c * 32 + hi * 8], &kT[cur][16 + r][c * 32 + 16 + hi * 8]);
      s0 = WMMA_F32_F16(aq[c], bk0, s0);
      s1 = WMMA_F32_F16(aq[c], bk1, s1);
    }
    const bool valid0 = mptr[k0 + r] != 0;
    const bool valid1 = mptr[k0 + 16 + r] != 0;

    float x0[8], x1[8], mnew[8], sc[8], p0[8], p1[8];
#pragma unroll
    for (int j = 0; j < 8; ++j) {
      x0[j] = valid0 ? s0[j] * 0.0625f : -1e9f;   // 1/sqrt(256), NEG_INF mask
      x1[j] = valid1 ? s1[j] * 0.0625f : -1e9f;
      float tm = fmaxf(x0[j], x1[j]);
#pragma unroll
      for (int off = 1; off < 16; off <<= 1) tm = fmaxf(tm, __shfl_xor(tm, off, 32));
      mnew[j] = fmaxf(mrow[j], tm);
      sc[j]   = __expf(mrow[j] - mnew[j]);
      p0[j]   = __expf(x0[j] - mnew[j]);
      p1[j]   = __expf(x1[j] - mnew[j]);
      float ps = p0[j] + p1[j];
#pragma unroll
      for (int off = 1; off < 16; off <<= 1) ps += __shfl_xor(ps, off, 32);
      lrow[j] = lrow[j] * sc[j] + ps;
      mrow[j] = mnew[j];
    }
#pragma unroll
    for (int nc = 0; nc < 16; ++nc)
#pragma unroll
      for (int j = 0; j < 8; ++j) o[nc][j] *= sc[j];

    // P (C-layout) -> per-wave LDS -> A-fragment (16x32 over keys)
#pragma unroll
    for (int j = 0; j < 8; ++j) {
      pT[wave][hi * 8 + j][r]      = (_Float16)p0[j];
      pT[wave][hi * 8 + j][16 + r] = (_Float16)p1[j];
    }
    __syncthreads();
    h16 pa = ld_h16_split(&pT[wave][r][hi * 8], &pT[wave][r][16 + hi * 8]);

    // o += P @ V from LDS-resident V
#pragma unroll
    for (int nc = 0; nc < 16; ++nc) {
      h16 bv = ld_h16_split(&vT[cur][nc * 16 + r][hi * 8], &vT[cur][nc * 16 + r][16 + hi * 8]);
      o[nc] = WMMA_F32_F16(pa, bv, o[nc]);
    }

    wait_async_stage();     // next tile staged
    __syncthreads();        // everyone done reading `cur`; safe to overwrite
  }

  _Float16* obase = attn + ((size_t)(b * SS + row0)) * HD + h * DD;
  float inv[8];
#pragma unroll
  for (int j = 0; j < 8; ++j) inv[j] = 1.f / lrow[j];
#pragma unroll
  for (int nc = 0; nc < 16; ++nc) {
    const int col = nc * 16 + r;
#pragma unroll
    for (int j = 0; j < 8; ++j)
      obase[(size_t)(hi * 8 + j) * HD + col] = (_Float16)(o[nc][j] * inv[j]);
  }
}

// ----------------------- output projection + bias + residual + LayerNorm
__global__ __launch_bounds__(256) void wo_ln_kernel(
    const _Float16* __restrict__ attn, // [MM, HD]
    const _Float16* __restrict__ Wo,   // [DD, HD]
    const float* __restrict__ wob, const float* __restrict__ Qin,
    const float* __restrict__ g, const float* __restrict__ lb,
    float* __restrict__ out) {
  __shared__ float tile[16 * DD];
  const int wave = threadIdx.x >> 5;
  const int lane = threadIdx.x & 31;
  const int r = lane & 15, hi = lane >> 4;
  const int row0 = blockIdx.x * 16;
  const int n0   = wave * 32;

  f8 acc0 = {}, acc1 = {};
  const _Float16* arow  = attn + (size_t)(row0 + r) * HD;
  const _Float16* b0row = Wo + (size_t)(n0 + r) * HD;
  const _Float16* b1row = Wo + (size_t)(n0 + 16 + r) * HD;
#pragma unroll 4
  for (int c = 0; c < HD / 32; ++c) {
    h16 a = ld_h16_split(arow + c * 32 + hi * 8, arow + c * 32 + 16 + hi * 8);
    acc0 = WMMA_F32_F16(a, ld_h16_split(b0row + c * 32 + hi * 8, b0row + c * 32 + 16 + hi * 8), acc0);
    acc1 = WMMA_F32_F16(a, ld_h16_split(b1row + c * 32 + hi * 8, b1row + c * 32 + 16 + hi * 8), acc1);
  }
  const int   c0 = n0 + r, c1 = n0 + 16 + r;
  const float bv0 = wob[c0], bv1 = wob[c1];
#pragma unroll
  for (int j = 0; j < 8; ++j) {
    const int row = hi * 8 + j;
    tile[row * DD + c0] = acc0[j] + bv0 + Qin[(size_t)(row0 + row) * DD + c0];
    tile[row * DD + c1] = acc1[j] + bv1 + Qin[(size_t)(row0 + row) * DD + c1];
  }
  __syncthreads();

  const int t = threadIdx.x;
  const int row = t >> 4;
  const int cc  = t & 15;
  float sum = 0.f, sq = 0.f;
#pragma unroll
  for (int j = 0; j < 16; ++j) {
    float v = tile[row * DD + cc + j * 16];
    sum += v; sq += v * v;
  }
#pragma unroll
  for (int off = 1; off < 16; off <<= 1) {
    sum += __shfl_xor(sum, off, 32);
    sq  += __shfl_xor(sq, off, 32);
  }
  const float mean = sum * (1.f / DD);
  const float var  = sq * (1.f / DD) - mean * mean;
  const float rs   = rsqrtf(var + 1e-5f);
#pragma unroll
  for (int j = 0; j < 16; ++j) {
    const int c = cc + j * 16;
    const float v = tile[row * DD + c];
    out[(size_t)(row0 + row) * DD + c] = (v - mean) * rs * g[c] + lb[c];
  }
}

// ---------------------------------------------------------------- launch
extern "C" void kernel_launch(void* const* d_in, const int* in_sizes, int n_in,
                              void* d_out, int out_size, void* d_ws, size_t ws_size,
                              hipStream_t stream) {
  (void)in_sizes; (void)n_in; (void)out_size; (void)ws_size;
  const float* Q   = (const float*)d_in[0];
  const int*  mask = (const int*)d_in[1];
  const float* WQw = (const float*)d_in[2];
  const float* WQb = (const float*)d_in[3];
  const float* WKw = (const float*)d_in[4];
  const float* WKb = (const float*)d_in[5];
  const float* WVw = (const float*)d_in[6];
  const float* WVb = (const float*)d_in[7];
  const float* WOw = (const float*)d_in[8];
  const float* WOb = (const float*)d_in[9];
  const float* lng = (const float*)d_in[10];
  const float* lnb = (const float*)d_in[11];
  float* out = (float*)d_out;

  char* ws = (char*)d_ws;
  size_t off = 0;
  auto take = [&](size_t bytes) {
    char* p = ws + off;
    off += (bytes + 255) & ~(size_t)255;
    return p;
  };
  _Float16* Xh  = (_Float16*)take((size_t)MM * DD * 2);
  _Float16* Wqh = (_Float16*)take((size_t)HD * DD * 2);
  _Float16* Wkh = (_Float16*)take((size_t)HD * DD * 2);
  _Float16* Wvh = (_Float16*)take((size_t)HD * DD * 2);
  _Float16* Woh = (_Float16*)take((size_t)DD * HD * 2);
  _Float16* qb  = (_Float16*)take((size_t)MM * HD * 2);
  _Float16* kbf = (_Float16*)take((size_t)MM * HD * 2);
  _Float16* vt  = (_Float16*)take((size_t)MM * HD * 2);
  _Float16* at  = (_Float16*)take((size_t)MM * HD * 2);

  auto cvt = [&](const float* s, _Float16* d, int n) {
    cvt_f16_kernel<<<(n + 255) / 256, 256, 0, stream>>>(s, d, n);
  };
  cvt(Q,   Xh,  MM * DD);
  cvt(WQw, Wqh, HD * DD);
  cvt(WKw, Wkh, HD * DD);
  cvt(WVw, Wvh, HD * DD);
  cvt(WOw, Woh, DD * HD);

  dim3 pgrid(MM / 16, HD / 128);
  proj_gemm_kernel<0><<<pgrid, 256, 0, stream>>>(Xh, Wqh, WQb, qb);
  proj_gemm_kernel<0><<<pgrid, 256, 0, stream>>>(Xh, Wkh, WKb, kbf);
  proj_gemm_kernel<1><<<pgrid, 256, 0, stream>>>(Xh, Wvh, WVb, vt);

  attn_kernel<<<dim3(SS / 64, HH, BB), 128, 0, stream>>>(qb, kbf, vt, mask, at);

  wo_ln_kernel<<<MM / 16, 256, 0, stream>>>(at, Woh, WOb, Q, lng, lnb, out);
}
